// CrossModalAttention_14912126452346
// MI455X (gfx1250) — matrix operand order repaired and checked by reading, not compiled, not run on hardware
//
#include <hip/hip_runtime.h>
#include <hip/hip_bf16.h>

#define N_VOX 8192
#define CCH   64

typedef __attribute__((ext_vector_type(16))) __bf16 v16bf;
typedef __attribute__((ext_vector_type(8)))  float  v8f;
typedef __attribute__((ext_vector_type(8)))  unsigned short vus8;

union FragB16 {
    v16bf bf;
    vus8  us[2];
    unsigned int u32[8];
};

static __device__ __forceinline__ unsigned short f2bf(float f) {
    unsigned u = __builtin_bit_cast(unsigned, f);
    unsigned r = u + 0x7FFFu + ((u >> 16) & 1u);   // round-to-nearest-even
    return (unsigned short)(r >> 16);
}

static __device__ __forceinline__ v8f wmma_bf16(v16bf a, v16bf b, v8f c) {
    return __builtin_amdgcn_wmma_f32_16x16x32_bf16(
        /*neg_a=*/false, a, /*neg_b=*/false, b,
        /*c_mod=*/(short)0, c, /*reuse_a=*/false, /*reuse_b=*/false);
}

// ---------------------------------------------------------------------------
// Kernel 1: per-voxel 1x1x1 conv projections (q,k,v) from one modality.
//   x   : (64, 8192) f32
//   qT  : (8192, 64) bf16, pre-scaled by qscale   (B-fragment friendly for S')
//   kT  : (8192, 64) bf16                          (A-fragment friendly for S')
//   vO  : (64, 8192) bf16                          (A-fragment friendly for PV)
// ---------------------------------------------------------------------------
__global__ void __launch_bounds__(256)
cma_proj_qkv(const float* __restrict__ x,
             const float* __restrict__ wq, const float* __restrict__ bq,
             const float* __restrict__ wk, const float* __restrict__ bk,
             const float* __restrict__ wv, const float* __restrict__ bv,
             unsigned short* __restrict__ qT,
             unsigned short* __restrict__ kT,
             unsigned short* __restrict__ vO,
             float qscale)
{
    __shared__ float xs[64][65];
    const int t  = threadIdx.x;
    const int n0 = blockIdx.x * 64;

    for (int idx = t; idx < 64 * 64; idx += 256) {
        int c = idx >> 6, n = idx & 63;
        xs[c][n] = x[c * N_VOX + n0 + n];
    }
    __syncthreads();

    const int nl = t & 63;
    const int ob = (t >> 6) * 16;           // uniform per wave -> scalar loads
    #pragma unroll 1
    for (int oo = 0; oo < 16; ++oo) {
        const int o = ob + oo;
        float aq = bq[o], ak = bk[o], av = bv[o];
        #pragma unroll
        for (int c = 0; c < 64; ++c) {
            float xv = xs[c][nl];
            aq = fmaf(wq[o * 64 + c], xv, aq);
            ak = fmaf(wk[o * 64 + c], xv, ak);
            av = fmaf(wv[o * 64 + c], xv, av);
        }
        const int n = n0 + nl;
        qT[n * 64 + o]    = f2bf(aq * qscale);
        kT[n * 64 + o]    = f2bf(ak);
        vO[o * N_VOX + n] = f2bf(av);
    }
}

// ---------------------------------------------------------------------------
// Kernel 2: flash attention with transposed S (S' = K x Q, shape j x i).
// One wave owns 16 query rows (i in lanes). Per 32-j step:
//   - 4 WMMAs: s'0,s'1 (16j x 16i each)  = K-tile(16j x 32c/32c) x Q(32c x 16i)
//   - softmax stats in-register (each lane holds 16 j-values of row i=lane%16),
//     one shfl_xor(16) for max, one for sum
//   - P B-fragment assembled from own regs + packed shfl_xor(16) exchange
//   - 4 WMMAs: o[ct] += V-tile(16c x 32j) x P(32j x 16i)   (O stored as c x i)
// ---------------------------------------------------------------------------
__global__ void __launch_bounds__(256)
cma_flash(const unsigned short* __restrict__ qT,
          const unsigned short* __restrict__ kT,
          const unsigned short* __restrict__ vM,
          float* __restrict__ fused, int colOff)
{
    const int t    = threadIdx.x;
    const int w    = t >> 5;
    const int lane = t & 31;
    const int hv   = lane >> 4;       // half-wave index
    const int lm   = lane & 15;
    const int i0   = (blockIdx.x * 8 + w) * 16;   // first query row of this wave

    // ---- Q B-fragments (held for whole loop): B[c,i], lane = column i.
    FragB16 bq0, bq1;
    {
        const unsigned short* qrow = qT + (i0 + lm) * 64 + 8 * hv;
        bq0.us[0] = *(const vus8*)(qrow);
        bq0.us[1] = *(const vus8*)(qrow + 16);
        bq1.us[0] = *(const vus8*)(qrow + 32);
        bq1.us[1] = *(const vus8*)(qrow + 48);
    }

    v8f o0 = {0,0,0,0,0,0,0,0}, o1 = {0,0,0,0,0,0,0,0};
    v8f o2 = {0,0,0,0,0,0,0,0}, o3 = {0,0,0,0,0,0,0,0};
    float mrun = -3.0e38f, lrun = 0.0f;

    for (int j0 = 0; j0 < N_VOX; j0 += 32) {
        // ---- K A-fragments: A[j,c] = kT[j*64 + c], row j in lane%16.
        FragB16 ak0l, ak0h, ak1l, ak1h;
        {
            const unsigned short* kr0 = kT + (j0 + lm) * 64;
            ak0l.us[0] = *(const vus8*)(kr0 + 8 * hv);
            ak0l.us[1] = *(const vus8*)(kr0 + 16 + 8 * hv);
            ak0h.us[0] = *(const vus8*)(kr0 + 32 + 8 * hv);
            ak0h.us[1] = *(const vus8*)(kr0 + 48 + 8 * hv);
            const unsigned short* kr1 = kr0 + 16 * 64;
            ak1l.us[0] = *(const vus8*)(kr1 + 8 * hv);
            ak1l.us[1] = *(const vus8*)(kr1 + 16 + 8 * hv);
            ak1h.us[0] = *(const vus8*)(kr1 + 32 + 8 * hv);
            ak1h.us[1] = *(const vus8*)(kr1 + 48 + 8 * hv);
        }

        // ---- S' tiles (already include softmax scale * log2e via qT).
        v8f s0 = {0,0,0,0,0,0,0,0}, s1 = {0,0,0,0,0,0,0,0};
        s0 = wmma_bf16(ak0l.bf, bq0.bf, s0);
        s0 = wmma_bf16(ak0h.bf, bq1.bf, s0);
        s1 = wmma_bf16(ak1l.bf, bq0.bf, s1);
        s1 = wmma_bf16(ak1h.bf, bq1.bf, s1);

        // ---- Online softmax: this lane's 16 values all belong to row i=lm.
        float mx = fmaxf(s0[0], s1[0]);
        #pragma unroll
        for (int r = 1; r < 8; ++r) mx = fmaxf(mx, fmaxf(s0[r], s1[r]));
        mx = fmaxf(mx, __shfl_xor(mx, 16));
        const float mnew = fmaxf(mrun, mx);
        const float corr = __builtin_amdgcn_exp2f(mrun - mnew);
        float p0[8], p1[8];
        float rsum = 0.0f;
        #pragma unroll
        for (int r = 0; r < 8; ++r) {
            p0[r] = __builtin_amdgcn_exp2f(s0[r] - mnew);
            p1[r] = __builtin_amdgcn_exp2f(s1[r] - mnew);
            rsum += p0[r] + p1[r];
        }
        rsum += __shfl_xor(rsum, 16);
        lrun = lrun * corr + rsum;
        mrun = mnew;
        o0 *= corr; o1 *= corr; o2 *= corr; o3 *= corr;

        // ---- Build P B-fragment (32j x 16i): half from own regs, half from
        //      the half-wave partner via one packed 4-dword exchange.
        unsigned int pk0[4], pk1[4];
        #pragma unroll
        for (int q = 0; q < 4; ++q) {
            pk0[q] = (unsigned)f2bf(p0[2*q]) | ((unsigned)f2bf(p0[2*q+1]) << 16);
            pk1[q] = (unsigned)f2bf(p1[2*q]) | ((unsigned)f2bf(p1[2*q+1]) << 16);
        }
        FragB16 bp;
        #pragma unroll
        for (int q = 0; q < 4; ++q) {
            unsigned int send = hv ? pk0[q] : pk1[q];
            unsigned int recv = __shfl_xor(send, 16);
            bp.u32[q]     = hv ? recv : pk0[q];
            bp.u32[4 + q] = hv ? pk1[q] : recv;
        }

        // ---- V A-fragments: A[c,j] = vM[c*N + j], row c in lane%16.
        FragB16 av0, av1, av2, av3;
        {
            const unsigned short* vb = vM + j0 + 16 * hv;
            av0.us[0] = *(const vus8*)(vb + (lm)      * N_VOX);
            av0.us[1] = *(const vus8*)(vb + (lm)      * N_VOX + 8);
            av1.us[0] = *(const vus8*)(vb + (16 + lm) * N_VOX);
            av1.us[1] = *(const vus8*)(vb + (16 + lm) * N_VOX + 8);
            av2.us[0] = *(const vus8*)(vb + (32 + lm) * N_VOX);
            av2.us[1] = *(const vus8*)(vb + (32 + lm) * N_VOX + 8);
            av3.us[0] = *(const vus8*)(vb + (48 + lm) * N_VOX);
            av3.us[1] = *(const vus8*)(vb + (48 + lm) * N_VOX + 8);
        }

        o0 = wmma_bf16(av0.bf, bp.bf, o0);
        o1 = wmma_bf16(av1.bf, bp.bf, o1);
        o2 = wmma_bf16(av2.bf, bp.bf, o2);
        o3 = wmma_bf16(av3.bf, bp.bf, o3);
    }

    // ---- Normalize and write (N,128) fused buffer: O is (c x i), lane = i.
    // o{ct}[r] is channel c = 16*ct + 8*hv + r of query row i0+lm: contiguous.
    const float inv = 1.0f / lrun;
    float* dst = fused + (i0 + lm) * 128 + colOff + 8 * hv;
    {
        v8f w0 = o0 * inv, w1 = o1 * inv, w2 = o2 * inv, w3 = o3 * inv;
        *(v8f*)(dst)      = w0;
        *(v8f*)(dst + 16) = w1;
        *(v8f*)(dst + 32) = w2;
        *(v8f*)(dst + 48) = w3;
    }
}

// ---------------------------------------------------------------------------
// Kernel 3: output projection  out[o,n] = bo[o] + sum_k wo[o,k]*fused[n,k]
// ---------------------------------------------------------------------------
__global__ void __launch_bounds__(256)
cma_outproj(const float* __restrict__ fused,
            const float* __restrict__ wo, const float* __restrict__ bo,
            float* __restrict__ out)
{
    __shared__ float fs[64][129];
    const int t  = threadIdx.x;
    const int n0 = blockIdx.x * 64;

    for (int idx = t; idx < 64 * 128; idx += 256) {
        int n = idx >> 7, k = idx & 127;
        fs[n][k] = fused[(n0 + n) * 128 + k];
    }
    __syncthreads();

    const int nl = t & 63;
    const int ob = (t >> 6) * 16;           // uniform per wave
    #pragma unroll 1
    for (int oo = 0; oo < 16; ++oo) {
        const int o = ob + oo;
        float acc = bo[o];
        #pragma unroll
        for (int k = 0; k < 128; ++k)
            acc = fmaf(wo[o * 128 + k], fs[nl][k], acc);
        out[o * N_VOX + n0 + nl] = acc;
    }
}

// ---------------------------------------------------------------------------
extern "C" void kernel_launch(void* const* d_in, const int* in_sizes, int n_in,
                              void* d_out, int out_size, void* d_ws, size_t ws_size,
                              hipStream_t stream)
{
    (void)in_sizes; (void)n_in; (void)out_size; (void)ws_size;

    const float* ct     = (const float*)d_in[0];
    const float* mri    = (const float*)d_in[1];
    const float* wq_ct  = (const float*)d_in[2];
    const float* bq_ct  = (const float*)d_in[3];
    const float* wk_mri = (const float*)d_in[4];
    const float* bk_mri = (const float*)d_in[5];
    const float* wv_mri = (const float*)d_in[6];
    const float* bv_mri = (const float*)d_in[7];
    const float* wq_mri = (const float*)d_in[8];
    const float* bq_mri = (const float*)d_in[9];
    const float* wk_ct  = (const float*)d_in[10];
    const float* bk_ct  = (const float*)d_in[11];
    const float* wv_ct  = (const float*)d_in[12];
    const float* bv_ct  = (const float*)d_in[13];
    const float* wo     = (const float*)d_in[14];
    const float* bo     = (const float*)d_in[15];

    char* ws = (char*)d_ws;
    const size_t matE = (size_t)N_VOX * CCH;     // elements per bf16 matrix
    unsigned short* qT_ct  = (unsigned short*)ws;
    unsigned short* kT_ct  = qT_ct  + matE;
    unsigned short* v_ct   = kT_ct  + matE;
    unsigned short* qT_mri = v_ct   + matE;
    unsigned short* kT_mri = qT_mri + matE;
    unsigned short* v_mri  = kT_mri + matE;
    float* fused = (float*)(ws + 6 * matE * sizeof(unsigned short));

    // scale = 1/sqrt(64); fold log2(e) so the flash kernel uses exp2 directly.
    const float QS = 0.125f * 1.44269504088896340736f;

    cma_proj_qkv<<<N_VOX / 64, 256, 0, stream>>>(
        ct, wq_ct, bq_ct, wk_ct, bk_ct, wv_ct, bv_ct,
        qT_ct, kT_ct, v_ct, QS);
    cma_proj_qkv<<<N_VOX / 64, 256, 0, stream>>>(
        mri, wq_mri, bq_mri, wk_mri, bk_mri, wv_mri, bv_mri,
        qT_mri, kT_mri, v_mri, QS);

    cma_flash<<<N_VOX / 128, 256, 0, stream>>>(qT_ct,  kT_mri, v_mri, fused, 0);
    cma_flash<<<N_VOX / 128, 256, 0, stream>>>(qT_mri, kT_ct,  v_ct,  fused, 64);

    cma_outproj<<<N_VOX / 64, 256, 0, stream>>>(fused, wo, bo, (float*)d_out);
}
